// GCN5_MN_NEG_67980742361104
// MI455X (gfx1250) — matrix operand order repaired and checked by reading, not compiled, not run on hardware
//
#include <hip/hip_runtime.h>
#include <hip/hip_bf16.h>

typedef __attribute__((ext_vector_type(16))) _Float16 v16h;
typedef __attribute__((ext_vector_type(8)))  _Float16 v8h;
typedef __attribute__((ext_vector_type(8)))  float    v8f;

#define HDIM 128

// ---------------- utility ----------------
__global__ void k_zero(float* __restrict__ p, long n) {
  long i = (long)blockIdx.x * blockDim.x + threadIdx.x;
  if (i < n) p[i] = 0.f;
}

// ---------------- degree build ----------------
__global__ void k_degrees(const int* __restrict__ src, const int* __restrict__ dst,
                          float* __restrict__ deg_out, float* __restrict__ deg_in, int E) {
  int e = blockIdx.x * blockDim.x + threadIdx.x;
  if (e < E) {
    atomicAdd(&deg_out[src[e]], 1.f);
    atomicAdd(&deg_in [dst[e]], 1.f);
  }
}

// norms + 8-wide input features from in-degree
__global__ void k_node_init(const float* __restrict__ deg_in, const float* __restrict__ deg_out,
                            float* __restrict__ norm_src, float* __restrict__ norm_dst,
                            float* __restrict__ h0, int N, int Npad) {
  int n = blockIdx.x * blockDim.x + threadIdx.x;
  if (n >= Npad) return;
  float* h = h0 + (size_t)n * 8;
  if (n < N) {
    float di = deg_in[n], dou = deg_out[n];
    norm_src[n] = rsqrtf(fmaxf(dou, 1.f));
    norm_dst[n] = rsqrtf(fmaxf(di, 1.f));
    float h1 = di;
    float h2 = (di > 3.f) ? 1.f : 0.f;
    float h3 = 3.f / di;                 // every node has in-degree >= 1 by construction
    float h4 = (di > 4.f) ? 1.f : 0.f;
    h[0] = h1;  h[1] = h2;  h[2] = h3;  h[3] = h4;
    h[4] = -h1; h[5] = -h2; h[6] = -h3; h[7] = -h4;
  } else {
    norm_src[n] = 0.f; norm_dst[n] = 0.f;
    for (int i = 0; i < 8; ++i) h[i] = 0.f;
  }
}

// ---------------- sparse aggregation: agg[dst] += h[src] * norm_src[src] ----------------
__global__ void k_scatter(const float* __restrict__ h, const int* __restrict__ src,
                          const int* __restrict__ dst, const float* __restrict__ norm_src,
                          float* __restrict__ agg, int E, int CH) {
  const int chunks = CH >> 2;
  long idx = (long)blockIdx.x * blockDim.x + threadIdx.x;
  long e = idx / chunks;
  int  c = (int)(idx - e * chunks) * 4;
  if (e >= E) return;
  int s = src[e], d = dst[e];
  float ns = norm_src[s];
  const float4 v = *reinterpret_cast<const float4*>(h + (size_t)s * CH + c);
  float* a = agg + (size_t)d * CH + c;
  atomicAdd(a + 0, v.x * ns);
  atomicAdd(a + 1, v.y * ns);
  atomicAdd(a + 2, v.z * ns);
  atomicAdd(a + 3, v.w * ns);
}

// ---------------- f32 agg * norm_dst -> f16 A operand (K padded, zero-filled) ----------------
__global__ void k_to_f16_scaled(const float* __restrict__ agg, const float* __restrict__ norm_dst,
                                _Float16* __restrict__ a16, int N, int Npad, int Kin, int K) {
  long idx = (long)blockIdx.x * blockDim.x + threadIdx.x;
  long total = (long)Npad * K;
  if (idx >= total) return;
  int n = (int)(idx / K);
  int k = (int)(idx - (long)n * K);
  float v = 0.f;
  if (n < N && k < Kin) v = agg[(size_t)n * Kin + k] * norm_dst[n];
  a16[idx] = (_Float16)v;
}

// ---------------- WMMA GEMM: out = relu(A16[Npad,K] @ W[KinW,128] + bias), fused epilogue ----
// K compile-time -> all loops unroll, fragments statically register-allocated.
// block = 256 threads = 8 waves; covers 64 rows x 128 cols.
// wave w: row tile = w&3, column group = w>>2 (4 x 16-col tiles each).
template <int K>
__launch_bounds__(256)
__global__ void k_gemm_wmma(const _Float16* __restrict__ A16,
                            const float* __restrict__ W, int KinW,
                            const float* __restrict__ bias,
                            float* __restrict__ out, int Nvalid) {
  constexpr int LDK    = K + 8;                 // padded LDS stride (bank rotate)
  constexpr int KSTEPS = K / 32;
  __shared__ __align__(16) _Float16 ldsWt[HDIM * LDK];  // W^T, [col][k]

  const int tid  = threadIdx.x;
  const int lane = tid & 31;
  const int wave = tid >> 5;

  // cooperative W^T load into LDS with f32->f16 convert + K zero-padding
  for (int i = tid; i < HDIM * K; i += 256) {
    int col = i / K;
    int k   = i - col * K;
    float v = (k < KinW) ? W[(size_t)k * HDIM + col] : 0.f;
    ldsWt[col * LDK + k] = (_Float16)v;
  }
  __syncthreads();

  const int rowbase = blockIdx.x * 64 + (wave & 3) * 16;
  const int colgrp  = wave >> 2;                // 0 or 1
  const int arow    = rowbase + (lane & 15);
  const int akb     = (lane >> 4) * 8;          // A 16-bit layout: lanes 16-31 start at K=8
  const int rowoff  = (lane & 16) ? 8 : 0;      // C layout: lanes 16-31 hold M = v+8
  const bool full   = (rowbase + 16 <= Nvalid); // wave-uniform: tail tile only in last block

  v16h a[KSTEPS];
  #pragma unroll
  for (int s = 0; s < KSTEPS; ++s) {
    const _Float16* ap = A16 + (size_t)arow * K + s * 32 + akb;
    v8h lo = *reinterpret_cast<const v8h*>(ap);        // K = akb .. akb+7
    v8h hi = *reinterpret_cast<const v8h*>(ap + 16);   // K = akb+16 .. akb+23
    #pragma unroll
    for (int i = 0; i < 8; ++i) { a[s][i] = lo[i]; a[s][i + 8] = hi[i]; }
  }

  #pragma unroll
  for (int ctl = 0; ctl < 4; ++ctl) {
    const int colbase = (colgrp * 4 + ctl) * 16;
    const int bcol    = colbase + (lane & 15);
    const _Float16* bp0 = ldsWt + bcol * LDK + (lane >> 4) * 8;   // B: lanes 16-31 -> K+8
    v8f acc = {};
    #pragma unroll
    for (int s = 0; s < KSTEPS; ++s) {
      const _Float16* bp = bp0 + s * 32;
      v8h blo = *reinterpret_cast<const v8h*>(bp);
      v8h bhi = *reinterpret_cast<const v8h*>(bp + 16);
      v16h b;
      #pragma unroll
      for (int i = 0; i < 8; ++i) { b[i] = blo[i]; b[i + 8] = bhi[i]; }
      acc = __builtin_amdgcn_wmma_f32_16x16x32_f16(false, a[s], false, b,
                                                   (short)0, acc, false, false);
    }
    const float bv  = bias[bcol];
    const int   row0 = rowbase + rowoff;
    float* outp = out + (size_t)row0 * HDIM + bcol;
    if (full) {
      // branch-free common path: 8 stores with constant immediate offsets
      #pragma unroll
      for (int v = 0; v < 8; ++v)
        outp[(size_t)v * HDIM] = fmaxf(acc[v] + bv, 0.f);
    } else {
      #pragma unroll
      for (int v = 0; v < 8; ++v)
        if (row0 + v < Nvalid)
          outp[(size_t)v * HDIM] = fmaxf(acc[v] + bv, 0.f);
    }
  }
}

// ---------------- graph mean pooling (graphs are contiguous 'per'-node blocks) -------------
__global__ void k_pool(const float* __restrict__ hco, float* __restrict__ emb, int per) {
  int g = blockIdx.x, c = threadIdx.x;   // 128 threads
  const float* p = hco + (size_t)g * per * HDIM + c;
  float s = 0.f;
  for (int n = 0; n < per; ++n) s += p[(size_t)n * HDIM];
  emb[g * HDIM + c] = s / (float)per;
}

// ---------------- readout MLP: relu(emb@lw1+lb1) @ lw2 + lb2 -> sigmoid --------------------
__global__ void k_readout(const float* __restrict__ emb,
                          const float* __restrict__ lw1, const float* __restrict__ lb1,
                          const float* __restrict__ lw2, const float* __restrict__ lb2,
                          float* __restrict__ pred) {
  __shared__ float red[64];
  int g = blockIdx.x, j = threadIdx.x;   // 64 threads
  const float* e = emb + g * HDIM;
  float s = lb1[j];
  for (int c = 0; c < HDIM; ++c) s += e[c] * lw1[c * 64 + j];
  s = fmaxf(s, 0.f);
  red[j] = s * lw2[j];
  __syncthreads();
  for (int off = 32; off > 0; off >>= 1) {
    if (j < off) red[j] += red[j + off];
    __syncthreads();
  }
  if (j == 0) pred[g] = 1.f / (1.f + expf(-(red[0] + lb2[0])));
}

// ---------------- host-side launcher ----------------
extern "C" void kernel_launch(void* const* d_in, const int* in_sizes, int n_in,
                              void* d_out, int out_size, void* d_ws, size_t ws_size,
                              hipStream_t stream) {
  const int*   src = (const int*)d_in[0];
  const int*   dst = (const int*)d_in[1];
  const float* W1  = (const float*)d_in[3];  const float* b1 = (const float*)d_in[4];
  const float* W2  = (const float*)d_in[5];  const float* b2 = (const float*)d_in[6];
  const float* W3  = (const float*)d_in[7];  const float* b3 = (const float*)d_in[8];
  const float* W4  = (const float*)d_in[9];  const float* b4 = (const float*)d_in[10];
  const float* W5  = (const float*)d_in[11]; const float* b5 = (const float*)d_in[12];
  const float* lw1 = (const float*)d_in[13]; const float* lb1 = (const float*)d_in[14];
  const float* lw2 = (const float*)d_in[15]; const float* lb2 = (const float*)d_in[16];

  const int E    = in_sizes[0];
  const int N    = in_sizes[2];
  const int G    = (int)(((long)out_size - (long)N * HDIM) / (1 + HDIM));
  const int per  = N / G;
  const int Npad = (N + 63) & ~63;

  float* ws        = (float*)d_ws;
  float* deg_in    = ws;                              // Npad
  float* deg_out   = deg_in  + Npad;                  // Npad
  float* norm_src  = deg_out + Npad;                  // Npad
  float* norm_dst  = norm_src + Npad;                 // Npad
  float* h0        = norm_dst + Npad;                 // Npad*8
  float* agg       = h0   + (size_t)Npad * 8;         // Npad*128
  float* bufA      = agg  + (size_t)Npad * HDIM;      // Npad*128
  float* bufB      = bufA + (size_t)Npad * HDIM;      // Npad*128
  _Float16* a16    = (_Float16*)(bufB + (size_t)Npad * HDIM);   // Npad*128 f16

  float* outP = (float*)d_out;
  float* pred = outP;
  float* emb  = outP + G;
  float* hco  = outP + G + (size_t)G * HDIM;

  auto cdiv = [](long a, long b) { return (int)((a + b - 1) / b); };

  // degrees + norms + input features
  k_zero     <<<cdiv(2L * Npad, 256), 256, 0, stream>>>(deg_in, 2L * Npad);
  k_degrees  <<<cdiv(E, 256),         256, 0, stream>>>(src, dst, deg_out, deg_in, E);
  k_node_init<<<cdiv(Npad, 256),      256, 0, stream>>>(deg_in, deg_out, norm_src, norm_dst,
                                                        h0, N, Npad);

  // layer 1: K padded 8 -> 32
  k_zero         <<<cdiv((long)Npad * 8, 256),  256, 0, stream>>>(agg, (long)Npad * 8);
  k_scatter      <<<cdiv((long)E * 2, 256),     256, 0, stream>>>(h0, src, dst, norm_src, agg, E, 8);
  k_to_f16_scaled<<<cdiv((long)Npad * 32, 256), 256, 0, stream>>>(agg, norm_dst, a16, N, Npad, 8, 32);
  k_gemm_wmma<32><<<Npad / 64, 256, 0, stream>>>(a16, W1, 8, b1, bufA, Npad);

  // layers 2..5
  const float* Ws[4]  = {W2, W3, W4, W5};
  const float* bs[4]  = {b2, b3, b4, b5};
  const float* ins[4] = {bufA, bufB, bufA, bufB};
  float*       outs[4]= {bufB, bufA, bufB, hco};
  for (int l = 0; l < 4; ++l) {
    k_zero         <<<cdiv((long)Npad * HDIM, 256),     256, 0, stream>>>(agg, (long)Npad * HDIM);
    k_scatter      <<<cdiv((long)E * (HDIM / 4), 256),  256, 0, stream>>>(ins[l], src, dst,
                                                                          norm_src, agg, E, HDIM);
    k_to_f16_scaled<<<cdiv((long)Npad * HDIM, 256),     256, 0, stream>>>(agg, norm_dst, a16,
                                                                          N, Npad, HDIM, HDIM);
    k_gemm_wmma<HDIM><<<Npad / 64, 256, 0, stream>>>(a16, Ws[l], HDIM, bs[l],
                                                     outs[l], (l == 3) ? N : Npad);
  }

  // pooling + readout
  k_pool   <<<G, HDIM, 0, stream>>>(hco, emb, per);
  k_readout<<<G, 64,   0, stream>>>(emb, lw1, lb1, lw2, lb2, pred);
}
